// MultiHeadAttention_4964982194257
// MI455X (gfx1250) — matrix operand order repaired and checked by reading, not compiled, not run on hardware
//
#include <hip/hip_runtime.h>
#include <math.h>

// ---------------------------------------------------------------------------
// Problem constants (from the reference)
// ---------------------------------------------------------------------------
#define BATCH     2
#define SEQ       2048
#define HIDDEN_D  2048
#define NHEADS    32
#define KVHEADS   8
#define HEADDIM   64
#define MROWS     (BATCH * SEQ)          // 4096
#define GROUPS    (NHEADS / KVHEADS)     // 4

typedef __attribute__((ext_vector_type(16))) __bf16 v16bf;
typedef __attribute__((ext_vector_type(8)))  float  v8f;

union Frag16 {
    v16bf v;
    uint4 u2[2];
    unsigned short s[16];
};

__device__ __forceinline__ unsigned short f2bf(float f) {
    union { float f; unsigned int u; } a;
    a.f = f;
    unsigned int r = a.u + 0x7FFFu + ((a.u >> 16) & 1u);   // round-to-nearest-even
    return (unsigned short)(r >> 16);
}

// A-fragment (16x32 bf16): lane row = lane&15, chunks at K = half*8, 16+half*8
// `ofs` must be a compile-time-foldable element offset so it lands in the
// VMEM instruction's 24-bit immediate instead of a live VGPR pair.
__device__ __forceinline__ void loadA(Frag16& f, const unsigned short* base,
                                      int ofs, int half) {
    f.u2[0] = *(const uint4*)(base + ofs + half * 8);
    f.u2[1] = *(const uint4*)(base + ofs + 16 + half * 8);
}
// B-fragment (32x16 bf16): lane col = lane&15, chunks at K = half*8, 16+half*8
__device__ __forceinline__ void loadB(Frag16& f, const unsigned short* base,
                                      int ofs, int half) {
    f.u2[0] = *(const uint4*)(base + ofs + half * 8);
    f.u2[1] = *(const uint4*)(base + ofs + 16 + half * 8);
}

// ---------------------------------------------------------------------------
// 1) f32 -> bf16 cast, 4 elements/thread
// ---------------------------------------------------------------------------
__global__ void cast_f32x4_bf16_kernel(const float4* __restrict__ src,
                                       uint2* __restrict__ dst, int n4) {
    int i = blockIdx.x * blockDim.x + threadIdx.x;
    if (i >= n4) return;
    float4 f = src[i];
    uint2 p;
    p.x = (unsigned int)f2bf(f.x) | ((unsigned int)f2bf(f.y) << 16);
    p.y = (unsigned int)f2bf(f.z) | ((unsigned int)f2bf(f.w) << 16);
    dst[i] = p;
}

// ---------------------------------------------------------------------------
// 2) GEMM: C[m,n] = sum_k A[m,k] * W[n,k]
//    A bf16 [M,K] rowmajor, W bf16 [N,K] rowmajor, C f32 [M,N].
//    K is a template constant (2048 everywhere here) so the inter-row strides
//    (16*K elements = 64 KB) become VMEM instruction immediates: only TWO
//    base pointers stay live.  One wave -> 32x64 tile (8 accumulators),
//    register double-buffered over K.  __launch_bounds__(256,1) lifts the
//    VGPR cap so the ~180-register working set does not spill.
// ---------------------------------------------------------------------------
template <int K>
__global__ __launch_bounds__(256, 1)
void gemm_nt_bf16_kernel(const unsigned short* __restrict__ A,
                         const unsigned short* __restrict__ W,
                         float* __restrict__ C,
                         int M, int N) {
    const int lane = threadIdx.x & 31;
    const int wv   = threadIdx.x >> 5;
    const int n16  = lane & 15;
    const int half = lane >> 4;

    const int tilesN = N >> 6;                 // 64-wide wave tiles in N
    const int waveId = blockIdx.x * 8 + wv;
    const int mt = waveId / tilesN;            // 32-row tiles in M
    const int nt = waveId % tilesN;
    if (mt * 32 >= M) return;

    // Two live base pointers; all other rows are immediate offsets.
    const unsigned short* abase = A + (size_t)(mt * 32 + n16) * K;
    const unsigned short* wbase = W + (size_t)(nt * 64 + n16) * K;

    v8f acc0[4] = {};
    v8f acc1[4] = {};

    Frag16 a0, a1, bw[4];
    loadA(a0, abase, 0, half);
    loadA(a1, abase, 16 * K, half);            // +16 rows -> 64 KB immediate
#pragma unroll
    for (int j = 0; j < 4; ++j) loadB(bw[j], wbase, j * 16 * K, half);

    for (int k0 = 0; k0 < K; k0 += 32) {
        // issue next iteration's loads first (wrap on last iter, result unused)
        const int kn = (k0 + 32 < K) ? k0 + 32 : 0;
        Frag16 a0n, a1n, bwn[4];
        loadA(a0n, abase, kn, half);
        loadA(a1n, abase, 16 * K + kn, half);
        __builtin_prefetch(abase + k0 + 512, 0, 1);
#pragma unroll
        for (int j = 0; j < 4; ++j) loadB(bwn[j], wbase, j * 16 * K + kn, half);

        // 8 WMMAs on the already-resident fragments
#pragma unroll
        for (int j = 0; j < 4; ++j) {
            acc0[j] = __builtin_amdgcn_wmma_f32_16x16x32_bf16(
                false, a0.v, false, bw[j].v, (short)0, acc0[j], false, false);
            acc1[j] = __builtin_amdgcn_wmma_f32_16x16x32_bf16(
                false, a1.v, false, bw[j].v, (short)0, acc1[j], false, false);
        }

        a0 = a0n; a1 = a1n;
#pragma unroll
        for (int j = 0; j < 4; ++j) bw[j] = bwn[j];
    }

#pragma unroll
    for (int j = 0; j < 4; ++j) {
#pragma unroll
        for (int v = 0; v < 8; ++v) {
            const int col = nt * 64 + j * 16 + n16;
            C[(size_t)(mt * 32 + half * 8 + v) * N + col]      = acc0[j][v];
            C[(size_t)(mt * 32 + 16 + half * 8 + v) * N + col] = acc1[j][v];
        }
    }
}

// ---------------------------------------------------------------------------
// 3) YaRN RoPE + layout pack:
//    qf f32 [b,s,32*64]  -> qb  bf16 [b,h,s,64]   (rotated, 1/sqrt(64) folded)
//    kf f32 [b,s, 8*64]  -> kbb bf16 [b,g,s,64]   (rotated)
//    vf f32 [b,s, 8*64]  -> vtb bf16 [b,g,64,s]   (transposed, no rotation)
//    one thread per (b, s, freq-index i in 0..31)
// ---------------------------------------------------------------------------
__device__ __forceinline__ float yarn_corr(float nrot) {
    return 64.0f * logf(4096.0f / (nrot * 6.28318530718f)) /
           (2.0f * logf(10000.0f));
}

__global__ void rope_pack_kernel(const float* __restrict__ qf,
                                 const float* __restrict__ kf,
                                 const float* __restrict__ vf,
                                 unsigned short* __restrict__ qb,
                                 unsigned short* __restrict__ kbb,
                                 unsigned short* __restrict__ vtb) {
    int idx = blockIdx.x * blockDim.x + threadIdx.x;
    if (idx >= BATCH * SEQ * 32) return;
    const int i = idx & 31;
    const int s = (idx >> 5) & (SEQ - 1);
    const int b = idx >> 16;                       // 5 + 11 bits

    // YaRN-corrected inverse frequency
    const float pf      = powf(10000.0f, (float)i / 32.0f);
    const float inv_ex  = 1.0f / pf;
    const float inv_in  = 1.0f / (195.3f * pf);
    const float low     = fmaxf(floorf(yarn_corr(32.0f)), 0.0f);
    const float high    = fminf(ceilf(yarn_corr(1.0f)), 31.0f);
    float ramp = ((float)i - low) / fmaxf(high - low, 0.001f);
    ramp = fminf(fmaxf(ramp, 0.0f), 1.0f);
    const float extrap  = 1.0f - ramp;
    const float invfreq = inv_in * (1.0f - extrap) + inv_ex * extrap;
    const float mscale  = 0.1f * logf(195.3f) + 1.0f;
    const float fr = (float)s * invfreq;
    const float c  = cosf(fr) * mscale;
    const float sn = sinf(fr) * mscale;
    const float qsc = 0.125f;                      // 64^-0.5 folded into Q

    // Q: 32 heads
#pragma unroll 4
    for (int h = 0; h < NHEADS; ++h) {
        const float* qr = qf + ((size_t)b * SEQ + s) * HIDDEN_D + h * HEADDIM;
        const float xr = qr[2 * i], xi = qr[2 * i + 1];
        unsigned short* qd =
            qb + (((size_t)(b * NHEADS + h)) * SEQ + s) * HEADDIM;
        qd[2 * i]     = f2bf((xr * c - xi * sn) * qsc);
        qd[2 * i + 1] = f2bf((xr * sn + xi * c) * qsc);
    }
    // K + V: 8 kv heads
#pragma unroll 4
    for (int g = 0; g < KVHEADS; ++g) {
        const float* kr = kf + ((size_t)b * SEQ + s) * (KVHEADS * HEADDIM) + g * HEADDIM;
        const float xr = kr[2 * i], xi = kr[2 * i + 1];
        unsigned short* kd =
            kbb + (((size_t)(b * KVHEADS + g)) * SEQ + s) * HEADDIM;
        kd[2 * i]     = f2bf(xr * c - xi * sn);
        kd[2 * i + 1] = f2bf(xr * sn + xi * c);

        const float* vr = vf + ((size_t)b * SEQ + s) * (KVHEADS * HEADDIM) + g * HEADDIM;
        unsigned short* vt =
            vtb + ((size_t)(b * KVHEADS + g) * HEADDIM) * SEQ;
        vt[(size_t)(2 * i) * SEQ + s]     = f2bf(vr[2 * i]);
        vt[(size_t)(2 * i + 1) * SEQ + s] = f2bf(vr[2 * i + 1]);
    }
}

// ---------------------------------------------------------------------------
// 4) Flash attention, one wave per (b, h, 16-query-row tile).
//    Scores: Q[16x64] x K^T via 2 x wmma per 16-key tile (scale pre-folded).
//    Online softmax in f32 C-fragment layout (row = half*8+v, col = lane&15).
//    Row-MAX uses 16-lane shuffles; row-SUM is computed on the matrix core:
//    a 5th accumulator accL = WMMA(P, ones) holds the running softmax
//    denominator, rescaled by the same alpha as the output accumulators, and
//    every lane ends up holding its row's denominator (all columns of P*1
//    are identical) -- no final broadcast needed.
// ---------------------------------------------------------------------------
__global__ __launch_bounds__(256, 1)
void flash_attn_kernel(const unsigned short* __restrict__ qb,
                       const unsigned short* __restrict__ kbb,
                       const unsigned short* __restrict__ vtb,
                       unsigned short* __restrict__ ctx) {
    __shared__ __align__(16) unsigned short Plds[8][16][40];   // 10 KB / block

    const int lane = threadIdx.x & 31;
    const int wv   = threadIdx.x >> 5;
    const int n16  = lane & 15;
    const int half = lane >> 4;

    const int waveId = blockIdx.x * 8 + wv;       // 8192 total
    const int qt  = waveId & 127;                 // SEQ/16
    const int h   = (waveId >> 7) & (NHEADS - 1);
    const int b   = waveId >> 12;
    const int kvh = h >> 2;                       // GQA: 4 q-heads per kv head
    const int qs  = qt * 16;

    // Load the two Q A-fragments (d = 0..31 and 32..63) once.
    const unsigned short* qrow =
        qb + (((size_t)(b * NHEADS + h)) * SEQ + qs + n16) * HEADDIM;
    Frag16 aQ0, aQ1;
    loadA(aQ0, qrow, 0, half);
    loadA(aQ1, qrow, 32, half);

    // All-ones bf16 B fragment for the row-sum WMMA.
    Frag16 bOne;
#pragma unroll
    for (int e = 0; e < 16; ++e) bOne.s[e] = 0x3F80;   // bf16 1.0

    float mi[8];
#pragma unroll
    for (int v = 0; v < 8; ++v) mi[v] = -1e30f;
    v8f acc[4] = {};
    v8f accL = {};                                 // running softmax denominator

    // Per-lane base pointers; everything else is an immediate offset.
    const unsigned short* kptr =
        kbb + ((size_t)(b * KVHEADS + kvh) * SEQ + n16) * HEADDIM;
    const unsigned short* vptr =
        vtb + ((size_t)(b * KVHEADS + kvh) * HEADDIM + n16) * SEQ;

    // Preload K fragments for the first 32-key block: bK[t*2+dh]
    Frag16 bK[4];
#pragma unroll
    for (int t = 0; t < 2; ++t) {
        loadB(bK[t * 2 + 0], kptr, t * 16 * HEADDIM, half);
        loadB(bK[t * 2 + 1], kptr, t * 16 * HEADDIM + 32, half);
    }

    for (int kb0 = 0; kb0 < SEQ; kb0 += 32) {
        // ---- scores for two 16-key tiles (resident fragments) ----------
        v8f sc[2];
#pragma unroll
        for (int t = 0; t < 2; ++t) {
            v8f c = {};
            c = __builtin_amdgcn_wmma_f32_16x16x32_bf16(
                false, aQ0.v, false, bK[t * 2 + 0].v, (short)0, c, false, false);
            c = __builtin_amdgcn_wmma_f32_16x16x32_bf16(
                false, aQ1.v, false, bK[t * 2 + 1].v, (short)0, c, false, false);
            sc[t] = c;
        }

        // ---- issue V loads (this block) + K loads (next block) early ---
        Frag16 bV[4];
#pragma unroll
        for (int dt = 0; dt < 4; ++dt)
            loadB(bV[dt], vptr + kb0, dt * 16 * SEQ, half);

        const int kbn = (kb0 + 32 < SEQ) ? kb0 + 32 : 0;
        Frag16 bKn[4];
#pragma unroll
        for (int t = 0; t < 2; ++t) {
            loadB(bKn[t * 2 + 0], kptr + (size_t)kbn * HEADDIM,
                  t * 16 * HEADDIM, half);
            loadB(bKn[t * 2 + 1], kptr + (size_t)kbn * HEADDIM,
                  t * 16 * HEADDIM + 32, half);
        }

        // ---- online softmax: row max via shuffles, exp, rescale --------
#pragma unroll
        for (int v = 0; v < 8; ++v) {
            const float s0 = sc[0][v];
            const float s1 = sc[1][v];
            float m = fmaxf(s0, s1);
#pragma unroll
            for (int off = 1; off < 16; off <<= 1)
                m = fmaxf(m, __shfl_xor(m, off, 32));
            const float mt = fmaxf(mi[v], m);
            const float al = __expf(mi[v] - mt);
            const float e0 = __expf(s0 - mt);
            const float e1 = __expf(s1 - mt);
            mi[v] = mt;
            acc[0][v] *= al; acc[1][v] *= al;
            acc[2][v] *= al; acc[3][v] *= al;
            accL[v]   *= al;
            Plds[wv][half * 8 + v][n16]      = f2bf(e0);
            Plds[wv][half * 8 + v][16 + n16] = f2bf(e1);
        }

        // ---- P (C-layout) -> A-fragment via per-wave LDS tile ----------
        Frag16 aP;
        const unsigned short* pr = &Plds[wv][n16][0];
        aP.u2[0] = *(const uint4*)(pr + half * 8);
        aP.u2[1] = *(const uint4*)(pr + 16 + half * 8);

        // ---- accumulate P x V over the 4 d-tiles + P x 1 (row sums) ----
#pragma unroll
        for (int dt = 0; dt < 4; ++dt) {
            acc[dt] = __builtin_amdgcn_wmma_f32_16x16x32_bf16(
                false, aP.v, false, bV[dt].v, (short)0, acc[dt], false, false);
        }
        accL = __builtin_amdgcn_wmma_f32_16x16x32_bf16(
            false, aP.v, false, bOne.v, (short)0, accL, false, false);

#pragma unroll
        for (int q = 0; q < 4; ++q) bK[q] = bKn[q];
    }

    // ---- finalize: divide by row sum (resident in accL), store ctx -----
#pragma unroll
    for (int v = 0; v < 8; ++v) {
        const float inv = 1.0f / accL[v];
        unsigned short* crow =
            ctx + ((size_t)b * SEQ + qs + half * 8 + v) * HIDDEN_D + h * HEADDIM;
#pragma unroll
        for (int dt = 0; dt < 4; ++dt)
            crow[dt * 16 + n16] = f2bf(acc[dt][v] * inv);
    }
}

// ---------------------------------------------------------------------------
// Host-side launcher
// ---------------------------------------------------------------------------
extern "C" void kernel_launch(void* const* d_in, const int* in_sizes, int n_in,
                              void* d_out, int out_size, void* d_ws, size_t ws_size,
                              hipStream_t stream) {
    (void)in_sizes; (void)n_in; (void)out_size; (void)ws_size;

    const float* hs = (const float*)d_in[0];   // [B,S,2048]
    const float* wq = (const float*)d_in[1];   // [2048,2048]
    const float* wk = (const float*)d_in[2];   // [512,2048]
    const float* wv = (const float*)d_in[3];   // [512,2048]
    const float* wo = (const float*)d_in[4];   // [2048,2048]
    float* out = (float*)d_out;                // [B,S,2048] f32

    char* ws = (char*)d_ws;
    size_t off = 0;
    auto carve = [&](size_t bytes) -> void* {
        void* p = ws + off;
        off = (off + bytes + 255) & ~(size_t)255;
        return p;
    };

    unsigned short* hsb = (unsigned short*)carve((size_t)MROWS * HIDDEN_D * 2);
    unsigned short* wqb = (unsigned short*)carve((size_t)HIDDEN_D * HIDDEN_D * 2);
    unsigned short* wkb = (unsigned short*)carve((size_t)512 * HIDDEN_D * 2);
    unsigned short* wvb = (unsigned short*)carve((size_t)512 * HIDDEN_D * 2);
    unsigned short* wob = (unsigned short*)carve((size_t)HIDDEN_D * HIDDEN_D * 2);
    float* qf = (float*)carve((size_t)MROWS * HIDDEN_D * 4);
    float* kf = (float*)carve((size_t)MROWS * 512 * 4);
    float* vf = (float*)carve((size_t)MROWS * 512 * 4);
    unsigned short* qbb = (unsigned short*)carve((size_t)MROWS * HIDDEN_D * 2);
    unsigned short* kbb = (unsigned short*)carve((size_t)MROWS * 512 * 2);
    unsigned short* vtb = (unsigned short*)carve((size_t)MROWS * 512 * 2);
    // ctx (bf16, 16.8 MB) aliases the qf region (f32, 33.6 MB) — qf is dead
    // after rope_pack.
    unsigned short* ctx = (unsigned short*)qf;

    const int T = 256;

    // 1) casts (4 elements per thread)
    {
        int n4;
        n4 = MROWS * HIDDEN_D / 4;
        cast_f32x4_bf16_kernel<<<(n4 + T - 1) / T, T, 0, stream>>>(
            (const float4*)hs, (uint2*)hsb, n4);
        n4 = HIDDEN_D * HIDDEN_D / 4;
        cast_f32x4_bf16_kernel<<<(n4 + T - 1) / T, T, 0, stream>>>(
            (const float4*)wq, (uint2*)wqb, n4);
        n4 = 512 * HIDDEN_D / 4;
        cast_f32x4_bf16_kernel<<<(n4 + T - 1) / T, T, 0, stream>>>(
            (const float4*)wk, (uint2*)wkb, n4);
        cast_f32x4_bf16_kernel<<<(n4 + T - 1) / T, T, 0, stream>>>(
            (const float4*)wv, (uint2*)wvb, n4);
        n4 = HIDDEN_D * HIDDEN_D / 4;
        cast_f32x4_bf16_kernel<<<(n4 + T - 1) / T, T, 0, stream>>>(
            (const float4*)wo, (uint2*)wob, n4);
    }

    // 2) Q/K/V projections (X * W^T), WMMA bf16 -> f32, 32x64 wave tiles
    {
        int waveTiles, blocks;
        waveTiles = (MROWS / 32) * (HIDDEN_D / 64);          // 4096
        blocks = waveTiles / 8;
        gemm_nt_bf16_kernel<HIDDEN_D><<<blocks, T, 0, stream>>>(
            hsb, wqb, qf, MROWS, HIDDEN_D);
        waveTiles = (MROWS / 32) * (512 / 64);               // 1024
        blocks = waveTiles / 8;
        gemm_nt_bf16_kernel<HIDDEN_D><<<blocks, T, 0, stream>>>(
            hsb, wkb, kf, MROWS, 512);
        gemm_nt_bf16_kernel<HIDDEN_D><<<blocks, T, 0, stream>>>(
            hsb, wvb, vf, MROWS, 512);
    }

    // 3) YaRN RoPE + layout pack (qb [b,h,s,d], kbb [b,g,s,d], vtb [b,g,d,s])
    {
        int n = BATCH * SEQ * 32;                            // 131072
        rope_pack_kernel<<<(n + T - 1) / T, T, 0, stream>>>(qf, kf, vf,
                                                            qbb, kbb, vtb);
    }

    // 4) flash attention -> ctx bf16 [b, s, 2048]
    {
        int waveTiles = BATCH * NHEADS * (SEQ / 16);         // 8192
        flash_attn_kernel<<<waveTiles / 8, T, 0, stream>>>(qbb, kbb, vtb, ctx);
    }

    // 5) output projection: ctx * wo^T -> f32 d_out
    {
        int waveTiles = (MROWS / 32) * (HIDDEN_D / 64);      // 4096
        gemm_nt_bf16_kernel<HIDDEN_D><<<waveTiles / 8, T, 0, stream>>>(
            ctx, wob, out, MROWS, HIDDEN_D);
    }
}
